// DNN_65944927863183
// MI455X (gfx1250) — compile-verified, matrix operand order checked by reading, not run on hardware
//
#include <hip/hip_runtime.h>
#include <hip/hip_bf16.h>
#include <cmath>

// ---------------------------------------------------------------------------
// MI455X (gfx1250) fused 4-block MLP:  [LN -> Linear -> ELU] x3 -> LN -> Linear(1)
// bf16 WMMA for GEMMs, f32 LN/epilogue.  64 rows per block, 16 wave32.
// K-outer / N-inner register tiling: per K-step a wave loads 4 A-frags (LDS)
// and NTPW B-frags (global); each B-frag feeds 4 WMMAs, each A-frag NTPW WMMAs.
// Bounded live set (~150 VGPRs) -> no scratch spills (round-2 pathology).
// ---------------------------------------------------------------------------

typedef __attribute__((ext_vector_type(16))) __bf16 v16bf;
typedef __attribute__((ext_vector_type(8)))  float  v8f;

union Frag16 {
    v16bf v;
    uint4 u[2];
};

#define ROWS    64              // rows per workgroup (4 WMMA M-tiles)
#define THREADS 512             // 16 wave32
#define XS      784             // bf16 LDS stride for LN outputs (768 + 16 pad)
#define AS      520             // f32 LDS stride for GEMM outputs (512 + 8 pad)
#define SMEM_BYTES (ROWS * XS * 2 + ROWS * AS * 4)   // 100352 + 133120 = 233472 B

// bf16 weight workspace layout (elements)
#define W0_OFF 0
#define W1_OFF 393216           // 512*768
#define W2_OFF 524288           // +256*512
#define WTOT   557056           // +128*256

// ---------------------------------------------------------------------------
// One-time weight conversion f32 -> bf16 into d_ws (stays hot in 192MB L2).
// ---------------------------------------------------------------------------
__global__ void cvt_weights_bf16(const float* __restrict__ W0,
                                 const float* __restrict__ W1,
                                 const float* __restrict__ W2,
                                 __bf16* __restrict__ ws)
{
    int i = blockIdx.x * 256 + threadIdx.x;
    if (i < W1_OFF)       ws[i] = (__bf16)W0[i];
    else if (i < W2_OFF)  ws[i] = (__bf16)W1[i - W1_OFF];
    else if (i < WTOT)    ws[i] = (__bf16)W2[i - W2_OFF];
}

// ---------------------------------------------------------------------------
// One GEMM layer:  act[64 x DOUT] = elu( Xbf[64 x DIN] @ W^T + bias )
// A-frag (16x32 bf16): lane<16 -> row=lane,   K = {0..7, 16..23}
//                      lane>=16 -> row=lane-16, K = {8..15, 24..31}
// B-frag mirrors this with N; W is row-major over K so each lane's 16
// K-values form two contiguous 8-element runs.  C/D f32: VGPR g -> M = g + 8*(lane>=16), N=lane&15.
// ---------------------------------------------------------------------------
template<int DIN, int DOUT, bool DO_ELU>
__device__ __forceinline__ void gemm_tile(const __bf16* Xbf,
                                          const __bf16* __restrict__ W,
                                          const float* __restrict__ bias,
                                          float* act, int tid)
{
    constexpr int NT   = DOUT / 16;                  // N-tiles in layer
    constexpr int NTPW = (NT >= 16) ? (NT / 16) : 1; // N-tiles per wave (2,1,1)

    const int lane = tid & 31;
    const int wave = tid >> 5;          // 0..15
    const int half = lane >> 4;         // 0 | 1
    const int l16  = lane & 15;
    const int ksel = half ? 8 : 0;

    if (wave >= NT) return;             // wave-uniform; only hits layer DOUT=128

    const __bf16* abase = Xbf + l16 * XS + ksel;

    const __bf16* brow[NTPW];
    #pragma unroll
    for (int j = 0; j < NTPW; ++j)
        brow[j] = W + ((wave + 16 * j) * 16 + l16) * DIN + ksel;

    v8f acc[NTPW][4];
    #pragma unroll
    for (int j = 0; j < NTPW; ++j) {
        #pragma unroll
        for (int mt = 0; mt < 4; ++mt)
            acc[j][mt] = (v8f){};
    }

    #pragma unroll 2                    // modest unroll: no giant live ranges
    for (int kk = 0; kk < DIN; kk += 32) {
        if ((kk & 127) == 0) {          // speculative prefetch of weight stream
            #pragma unroll
            for (int j = 0; j < NTPW; ++j)
                __builtin_prefetch((const void*)(brow[j] + kk + 512), 0, 3);
        }
        Frag16 a[4];
        #pragma unroll
        for (int mt = 0; mt < 4; ++mt) {
            const __bf16* ap = abase + mt * (16 * XS) + kk;
            a[mt].u[0] = *(const uint4*)(ap);
            a[mt].u[1] = *(const uint4*)(ap + 16);
        }
        #pragma unroll
        for (int j = 0; j < NTPW; ++j) {
            Frag16 b;
            b.u[0] = *(const uint4*)(brow[j] + kk);
            b.u[1] = *(const uint4*)(brow[j] + kk + 16);
            #pragma unroll
            for (int mt = 0; mt < 4; ++mt)
                acc[j][mt] = __builtin_amdgcn_wmma_f32_16x16x32_bf16(
                    false, a[mt].v, false, b.v, (short)0, acc[j][mt], false, false);
        }
    }

    #pragma unroll
    for (int j = 0; j < NTPW; ++j) {
        const int n  = (wave + 16 * j) * 16 + l16;
        const float cb = bias[n];
        #pragma unroll
        for (int mt = 0; mt < 4; ++mt) {
            const int mbase = mt * 16 + half * 8;
            #pragma unroll
            for (int g = 0; g < 8; ++g) {
                float v = acc[j][mt][g] + cb;
                if (DO_ELU) v = (v > 0.0f) ? v : expm1f(v);
                act[(mbase + g) * AS + n] = v;
            }
        }
    }
}

// ---------------------------------------------------------------------------
// Hidden LayerNorm: f32 act (LDS) -> bf16 Xbf (LDS).  8 lanes per row.
// ---------------------------------------------------------------------------
template<int D>
__device__ __forceinline__ void ln_hidden(const float* act, __bf16* Xbf,
                                          const float* __restrict__ g,
                                          const float* __restrict__ b,
                                          int tid)
{
    const int row = tid >> 3;            // 0..63 with 512 threads
    const int sub = tid & 7;
    const float* ap = act + row * AS;
    float s = 0.f, ss = 0.f;
    #pragma unroll
    for (int j = 0; j < D / 8; ++j) {
        float x = ap[j * 8 + sub];
        s += x; ss += x * x;
    }
    #pragma unroll
    for (int m = 4; m; m >>= 1) {
        s  += __shfl_xor(s,  m, 8);
        ss += __shfl_xor(ss, m, 8);
    }
    const float mean = s * (1.0f / D);
    const float rs   = rsqrtf(ss * (1.0f / D) - mean * mean + 1e-5f);
    #pragma unroll
    for (int j = 0; j < D / 8; ++j) {
        const int k = j * 8 + sub;
        Xbf[row * XS + k] = (__bf16)((ap[k] - mean) * rs * g[k] + b[k]);
    }
}

// ---------------------------------------------------------------------------
// Fused MLP kernel: one block = 64 rows, 512 threads (16 wave32).
// ---------------------------------------------------------------------------
__global__ __launch_bounds__(THREADS)
void mlp4_kernel(const float* __restrict__ input,
                 const float* __restrict__ g0, const float* __restrict__ b0, const float* __restrict__ c0,
                 const float* __restrict__ g1, const float* __restrict__ b1, const float* __restrict__ c1,
                 const float* __restrict__ g2, const float* __restrict__ b2, const float* __restrict__ c2,
                 const float* __restrict__ g3, const float* __restrict__ b3,
                 const float* __restrict__ W3, const float* __restrict__ c3,
                 const __bf16* __restrict__ wbf,
                 float* __restrict__ out)
{
    extern __shared__ char smem[];
    __bf16* Xbf = (__bf16*)smem;                       // 64 x 784 bf16
    float*  act = (float*)(smem + ROWS * XS * 2);      // 64 x 520 f32
    const int tid = threadIdx.x;
    const size_t row0 = (size_t)blockIdx.x * ROWS;

    // ---- LN0: stream 768-wide rows from global once; 16 lanes per row ----
    #pragma unroll
    for (int rbase = 0; rbase < ROWS; rbase += 32) {
        const int row = rbase + (tid >> 4);            // 32 rows per pass
        const int sub = tid & 15;
        const float* xp = input + (row0 + row) * 768;
        float vals[48];
        float s = 0.f, ss = 0.f;
        #pragma unroll
        for (int j = 0; j < 48; ++j) {
            float x = xp[j * 16 + sub];                // coalesced across lanes
            vals[j] = x; s += x; ss += x * x;
        }
        #pragma unroll
        for (int m = 8; m; m >>= 1) {
            s  += __shfl_xor(s,  m, 16);
            ss += __shfl_xor(ss, m, 16);
        }
        const float mean = s * (1.0f / 768.0f);
        const float rs   = rsqrtf(ss * (1.0f / 768.0f) - mean * mean + 1e-5f);
        #pragma unroll
        for (int j = 0; j < 48; ++j) {
            const int k = j * 16 + sub;
            Xbf[row * XS + k] = (__bf16)((vals[j] - mean) * rs * g0[k] + b0[k]);
        }
    }
    __syncthreads();

    gemm_tile<768, 512, true>(Xbf, wbf + W0_OFF, c0, act, tid);
    __syncthreads();
    ln_hidden<512>(act, Xbf, g1, b1, tid);
    __syncthreads();
    gemm_tile<512, 256, true>(Xbf, wbf + W1_OFF, c1, act, tid);
    __syncthreads();
    ln_hidden<256>(act, Xbf, g2, b2, tid);
    __syncthreads();
    gemm_tile<256, 128, true>(Xbf, wbf + W2_OFF, c2, act, tid);
    __syncthreads();

    // ---- final block: LN(128) + dot with W3 in f32, one row per thread ----
    if (tid < ROWS) {
        const float* ap = act + tid * AS;
        float s = 0.f, ss = 0.f;
        #pragma unroll 8
        for (int k = 0; k < 128; ++k) { float x = ap[k]; s += x; ss += x * x; }
        const float mean = s * (1.0f / 128.0f);
        const float rs   = rsqrtf(ss * (1.0f / 128.0f) - mean * mean + 1e-5f);
        float dot = 0.f;
        #pragma unroll 8
        for (int k = 0; k < 128; ++k)
            dot += ((ap[k] - mean) * rs * g3[k] + b3[k]) * W3[k];
        out[row0 + tid] = dot + c3[0];
    }
}

// ---------------------------------------------------------------------------
// Launch
// ---------------------------------------------------------------------------
extern "C" void kernel_launch(void* const* d_in, const int* in_sizes, int n_in,
                              void* d_out, int out_size, void* d_ws, size_t ws_size,
                              hipStream_t stream)
{
    const float* input = (const float*)d_in[0];
    const float* g0 = (const float*)d_in[1];
    const float* b0 = (const float*)d_in[2];
    const float* W0 = (const float*)d_in[3];
    const float* c0 = (const float*)d_in[4];
    const float* g1 = (const float*)d_in[5];
    const float* b1 = (const float*)d_in[6];
    const float* W1 = (const float*)d_in[7];
    const float* c1 = (const float*)d_in[8];
    const float* g2 = (const float*)d_in[9];
    const float* b2 = (const float*)d_in[10];
    const float* W2 = (const float*)d_in[11];
    const float* c2 = (const float*)d_in[12];
    const float* g3 = (const float*)d_in[13];
    const float* b3 = (const float*)d_in[14];
    const float* W3 = (const float*)d_in[15];
    const float* c3 = (const float*)d_in[16];

    __bf16* wbf = (__bf16*)d_ws;

    // Convert weights to bf16 (tiny; stays in L2 for the whole run).
    cvt_weights_bf16<<<(WTOT + 255) / 256, 256, 0, stream>>>(W0, W1, W2, wbf);

    // Allow >64KB dynamic LDS (320KB/WGP on CDNA5).
    (void)hipFuncSetAttribute((const void*)mlp4_kernel,
                              hipFuncAttributeMaxDynamicSharedMemorySize,
                              SMEM_BYTES);

    const int nblocks = 262144 / ROWS;   // 4096
    mlp4_kernel<<<nblocks, THREADS, SMEM_BYTES, stream>>>(
        input, g0, b0, c0, g1, b1, c1, g2, b2, c2, g3, b3, W3, c3,
        wbf, (float*)d_out);
}
